// ObjectDetector_18751827214781
// MI455X (gfx1250) — matrix operand order repaired and checked by reading, not compiled, not run on hardware
//
#include <hip/hip_runtime.h>
#include <hip/hip_fp16.h>

typedef _Float16 v16h __attribute__((ext_vector_type(16)));
typedef _Float16 half8 __attribute__((ext_vector_type(8)));
typedef float    v8f  __attribute__((ext_vector_type(8)));

#define BB 16
#define NN 4096
#define CC 256
#define DD 128
#define SS 10
#define HID 256
#define TT 20
#define ITERS 3
#define LN_EPS 1e-5f
#define ATT_EPS 1e-8f
#define QK_SCALE 0.08838834764831845f   // 128^-0.5

// ---------------- workspace layout (bytes) ----------------
#define OFF_PACKWK  ((size_t)0)                      // 8*8*32*16 f16 = 64KB
#define OFF_PACKWV  ((size_t)65536)
#define OFF_K       ((size_t)131072)                 // 65536*128 f16 = 16MB
#define OFF_V       (OFF_K + (size_t)65536*128*2)
#define OFF_SLOTS   (OFF_V + (size_t)65536*128*2)    // 16*10*128 f32
#define OFF_Q       (OFF_SLOTS + (size_t)BB*SS*DD*4)
#define OFF_ATTN    (OFF_Q + (size_t)BB*SS*DD*4)     // 16*10*4096 f32

// ---------------- K0: repack Wk/Wv into wave32 B-fragment layout ----------------
// pack[kk][nt][lane][h] : lane&15 = col n, lane>>4 selects K octets (0..7,16..23 / 8..15,24..31)
__global__ void repack_weights(const float* __restrict__ Wk,
                               const float* __restrict__ Wv,
                               _Float16* __restrict__ packK,
                               _Float16* __restrict__ packV) {
    int tid = blockIdx.x * blockDim.x + threadIdx.x;
    if (tid >= 4096) return;
    const float* W = (tid < 2048) ? Wk : Wv;
    _Float16* P = (tid < 2048) ? packK : packV;
    int t = tid & 2047;
    int kk = t >> 8, nt = (t >> 5) & 7, lane = t & 31;
    int n  = lane & 15;
    int kh = (lane >> 4) * 8;
    _Float16* dst = P + (size_t)t * 16;
    #pragma unroll
    for (int h = 0; h < 16; ++h)
        dst[h] = (_Float16)W[(size_t)(kk * 32 + kh + h + (h & 8)) * DD + nt * 16 + n];
}

// ---------------- slots init ----------------
__global__ void init_slots(const float* __restrict__ mu, const float* __restrict__ lsig,
                           const float* __restrict__ noise, float* __restrict__ slots) {
    int i = blockIdx.x * blockDim.x + threadIdx.x;
    if (i >= BB * SS * DD) return;
    int d = i & (DD - 1);
    slots[i] = mu[d] + expf(lsig[d]) * noise[i];
}

// ---------------- K1: fused LN(x) + k = xln@Wk, v = xln@Wv via WMMA f16 ----------------
// 128 threads = 4 waves; each wave owns a 16-row tile (K=256 -> D=128 for k and v)
__global__ __launch_bounds__(128) void ln_project(
        const float* __restrict__ obs, const float* __restrict__ niw,
        const float* __restrict__ nib, const _Float16* __restrict__ packK,
        const _Float16* __restrict__ packV, _Float16* __restrict__ kout,
        _Float16* __restrict__ vout) {
    __shared__ _Float16 xln[4][16][CC];   // 32KB
    int wv = threadIdx.x >> 5;
    int lane = threadIdx.x & 31;
    int r0 = blockIdx.x * 64 + wv * 16;

    // ---- LayerNorm: 2 lanes per row, each lane covers 128 of 256 elements ----
    int rowl = lane >> 1;
    int half = lane & 1;
    const float* xp = obs + ((size_t)(r0 + rowl)) * CC + half * 128;
    float s = 0.f, sq = 0.f;
    #pragma unroll 8
    for (int j = 0; j < 128; j += 4) {
        float4 t = *(const float4*)(xp + j);
        s  += t.x + t.y + t.z + t.w;
        sq += t.x * t.x + t.y * t.y + t.z * t.z + t.w * t.w;
    }
    s  += __shfl_xor(s, 1, 32);
    sq += __shfl_xor(sq, 1, 32);
    float mean = s * (1.f / 256.f);
    float var  = sq * (1.f / 256.f) - mean * mean;
    float rstd = rsqrtf(var + LN_EPS);
    #pragma unroll 8
    for (int j = 0; j < 128; j += 4) {
        float4 t = *(const float4*)(xp + j);
        int c = half * 128 + j;
        xln[wv][rowl][c + 0] = (_Float16)((t.x - mean) * rstd * niw[c + 0] + nib[c + 0]);
        xln[wv][rowl][c + 1] = (_Float16)((t.y - mean) * rstd * niw[c + 1] + nib[c + 1]);
        xln[wv][rowl][c + 2] = (_Float16)((t.z - mean) * rstd * niw[c + 2] + nib[c + 2]);
        xln[wv][rowl][c + 3] = (_Float16)((t.w - mean) * rstd * niw[c + 3] + nib[c + 3]);
    }
    __syncthreads();

    // ---- GEMM: 16x128 output tile for both k and v, K=256 ----
    v8f acck[8], accv[8];
    #pragma unroll
    for (int i = 0; i < 8; ++i) { acck[i] = (v8f)0.f; accv[i] = (v8f)0.f; }
    const v16h* bk = (const v16h*)packK;
    const v16h* bv = (const v16h*)packV;
    int m = lane & 15;
    int ksel = lane >> 4;

    #pragma unroll
    for (int kk = 0; kk < 8; ++kk) {
        // A fragment: 16x32 f16 per ISA layout
        int base = kk * 32 + ksel * 8;
        half8 lo = *(const half8*)&xln[wv][m][base];
        half8 hi = *(const half8*)&xln[wv][m][base + 16];
        v16h a = __builtin_shufflevector(lo, hi, 0,1,2,3,4,5,6,7,8,9,10,11,12,13,14,15);
        #pragma unroll
        for (int nt = 0; nt < 8; ++nt) {
            v16h bK = bk[(size_t)(kk * 8 + nt) * 32 + lane];
            v16h bV = bv[(size_t)(kk * 8 + nt) * 32 + lane];
            acck[nt] = __builtin_amdgcn_wmma_f32_16x16x32_f16(
                false, a, false, bK, (short)0, acck[nt], false, false);
            accv[nt] = __builtin_amdgcn_wmma_f32_16x16x32_f16(
                false, a, false, bV, (short)0, accv[nt], false, false);
        }
    }

    // ---- store (f16) per 16x16 f32 C/D layout: vgpr r -> M=r (lanes 0-15) / M=8+r ----
    int n_ = lane & 15;
    int mh = ksel * 8;
    #pragma unroll
    for (int nt = 0; nt < 8; ++nt) {
        #pragma unroll
        for (int r = 0; r < 8; ++r) {
            size_t o = (size_t)(r0 + mh + r) * DD + nt * 16 + n_;
            kout[o] = (_Float16)acck[nt][r];
            vout[o] = (_Float16)accv[nt][r];
        }
    }
}

// ---------------- per-iteration: q = LN(slots) @ Wq ----------------
__global__ __launch_bounds__(128) void q_proj(
        const float* __restrict__ slots, const float* __restrict__ nsw,
        const float* __restrict__ nsb, const float* __restrict__ Wq,
        float* __restrict__ q) {
    __shared__ float sl[SS][DD];
    __shared__ float hln[SS][DD];
    __shared__ float red[DD];
    int b = blockIdx.x, tid = threadIdx.x;
    for (int s = 0; s < SS; ++s) sl[s][tid] = slots[(size_t)(b * SS + s) * DD + tid];
    __syncthreads();
    for (int s = 0; s < SS; ++s) {
        float x = sl[s][tid];
        red[tid] = x; __syncthreads();
        for (int off = 64; off > 0; off >>= 1) { if (tid < off) red[tid] += red[tid + off]; __syncthreads(); }
        float s1 = red[0]; __syncthreads();
        red[tid] = x * x; __syncthreads();
        for (int off = 64; off > 0; off >>= 1) { if (tid < off) red[tid] += red[tid + off]; __syncthreads(); }
        float s2 = red[0]; __syncthreads();
        float mean = s1 * (1.f / DD);
        float var  = s2 * (1.f / DD) - mean * mean;
        hln[s][tid] = (x - mean) * rsqrtf(var + LN_EPS) * nsw[tid] + nsb[tid];
    }
    __syncthreads();
    for (int s = 0; s < SS; ++s) {
        float acc = 0.f;
        #pragma unroll 4
        for (int j = 0; j < DD; ++j) acc += hln[s][j] * Wq[(size_t)j * DD + tid];
        q[(size_t)(b * SS + s) * DD + tid] = acc;
    }
}

// ---------------- per-iteration: logits + softmax over slots ----------------
__global__ __launch_bounds__(128) void attn_softmax(
        const float* __restrict__ q, const _Float16* __restrict__ k,
        float* __restrict__ attn) {
    __shared__ float qs[SS * DD];
    int b = blockIdx.y, tid = threadIdx.x;
    int n = blockIdx.x * 128 + tid;
    for (int i = tid; i < SS * DD; i += 128) qs[i] = q[(size_t)b * SS * DD + i];
    __syncthreads();
    const _Float16* kr = k + ((size_t)b * NN + n) * DD;
    float lg[SS];
    #pragma unroll
    for (int s = 0; s < SS; ++s) lg[s] = 0.f;
    for (int d0 = 0; d0 < DD; d0 += 8) {
        half8 kv8 = *(const half8*)(kr + d0);
        #pragma unroll
        for (int j = 0; j < 8; ++j) {
            float kv = (float)kv8[j];
            #pragma unroll
            for (int s = 0; s < SS; ++s) lg[s] += qs[s * DD + d0 + j] * kv;
        }
    }
    float mx = -3.0e38f;
    #pragma unroll
    for (int s = 0; s < SS; ++s) { lg[s] *= QK_SCALE; mx = fmaxf(mx, lg[s]); }
    float sum = 0.f;
    #pragma unroll
    for (int s = 0; s < SS; ++s) { lg[s] = expf(lg[s] - mx); sum += lg[s]; }
    float inv = 1.f / sum;
    #pragma unroll
    for (int s = 0; s < SS; ++s)
        attn[((size_t)(b * SS + s)) * NN + n] = lg[s] * inv;
}

// ---------------- per-iteration: updates + residual + LN + MLP ----------------
__global__ __launch_bounds__(128) void slot_update(
        const float* __restrict__ attn, const _Float16* __restrict__ v,
        const float* __restrict__ nmw, const float* __restrict__ nmb,
        const float* __restrict__ W1, const float* __restrict__ b1,
        const float* __restrict__ W2, const float* __restrict__ b2,
        float* __restrict__ slots) {
    __shared__ float abuf[SS][128];
    __shared__ float sl[SS][DD];
    __shared__ float hb[SS][DD];
    __shared__ float tb[SS * HID];
    __shared__ float red[DD];
    int b = blockIdx.x, tid = threadIdx.x;
    float acc[SS], asum[SS];
    #pragma unroll
    for (int s = 0; s < SS; ++s) { acc[s] = 0.f; asum[s] = 0.f; }

    for (int n0 = 0; n0 < NN; n0 += 128) {
        for (int s = 0; s < SS; ++s)
            abuf[s][tid] = attn[((size_t)(b * SS + s)) * NN + n0 + tid];
        __syncthreads();
        for (int j = 0; j < 128; ++j) {
            float vv = (float)v[((size_t)b * NN + n0 + j) * DD + tid];
            #pragma unroll
            for (int s = 0; s < SS; ++s) {
                float a = abuf[s][j];
                acc[s]  += a * vv;
                asum[s] += a;
            }
        }
        __syncthreads();
    }
    // slots = slots_prev + attn-weighted mean of v
    for (int s = 0; s < SS; ++s)
        sl[s][tid] = slots[(size_t)(b * SS + s) * DD + tid] + acc[s] / (asum[s] + ATT_EPS);
    __syncthreads();
    // LN
    for (int s = 0; s < SS; ++s) {
        float x = sl[s][tid];
        red[tid] = x; __syncthreads();
        for (int off = 64; off > 0; off >>= 1) { if (tid < off) red[tid] += red[tid + off]; __syncthreads(); }
        float s1 = red[0]; __syncthreads();
        red[tid] = x * x; __syncthreads();
        for (int off = 64; off > 0; off >>= 1) { if (tid < off) red[tid] += red[tid + off]; __syncthreads(); }
        float s2 = red[0]; __syncthreads();
        float mean = s1 * (1.f / DD);
        float var  = s2 * (1.f / DD) - mean * mean;
        hb[s][tid] = (x - mean) * rsqrtf(var + LN_EPS) * nmw[tid] + nmb[tid];
    }
    __syncthreads();
    // MLP layer 1 (128 -> 256) with ReLU
    for (int s = 0; s < SS; ++s) {
        #pragma unroll
        for (int half = 0; half < 2; ++half) {
            int jj = tid + half * 128;
            float a2 = b1[jj];
            #pragma unroll 4
            for (int d = 0; d < DD; ++d) a2 += hb[s][d] * W1[(size_t)d * HID + jj];
            tb[s * HID + jj] = fmaxf(a2, 0.f);
        }
    }
    __syncthreads();
    // MLP layer 2 (256 -> 128) + residual
    for (int s = 0; s < SS; ++s) {
        float o = b2[tid];
        #pragma unroll 4
        for (int j = 0; j < HID; ++j) o += tb[s * HID + j] * W2[(size_t)j * DD + tid];
        slots[(size_t)(b * SS + s) * DD + tid] = sl[s][tid] + o;
    }
}

// ---------------- output heads ----------------
__global__ __launch_bounds__(128) void heads(
        const float* __restrict__ slots,
        const float* __restrict__ pw1, const float* __restrict__ pb1,
        const float* __restrict__ pw2, const float* __restrict__ pb2,
        const float* __restrict__ tcw, const float* __restrict__ tcb,
        float* __restrict__ out) {
    __shared__ float sl[SS][DD];
    __shared__ float tb[SS * HID];
    int b = blockIdx.x, tid = threadIdx.x;
    for (int s = 0; s < SS; ++s) sl[s][tid] = slots[(size_t)(b * SS + s) * DD + tid];
    __syncthreads();
    for (int s = 0; s < SS; ++s) {
        #pragma unroll
        for (int half = 0; half < 2; ++half) {
            int jj = tid + half * 128;
            float a = pb1[jj];
            #pragma unroll 4
            for (int d = 0; d < DD; ++d) a += sl[s][d] * pw1[(size_t)d * HID + jj];
            tb[s * HID + jj] = fmaxf(a, 0.f);
        }
    }
    __syncthreads();
    for (int s = 0; s < SS; ++s) {
        float o = pb2[tid];
        #pragma unroll 4
        for (int j = 0; j < HID; ++j) o += tb[s * HID + j] * pw2[(size_t)j * DD + tid];
        out[(size_t)(b * SS + s) * DD + tid] = o;          // objects
    }
    if (tid < TT) {
        for (int s = 0; s < SS; ++s) {
            float o2 = tcb[tid];
            #pragma unroll 4
            for (int d = 0; d < DD; ++d) o2 += sl[s][d] * tcw[(size_t)d * TT + tid];
            out[(size_t)BB * SS * DD + (size_t)(b * SS + s) * TT + tid] = o2;   // object_types
        }
    }
}

extern "C" void kernel_launch(void* const* d_in, const int* in_sizes, int n_in,
                              void* d_out, int out_size, void* d_ws, size_t ws_size,
                              hipStream_t stream) {
    const float* obs   = (const float*)d_in[0];
    const float* noise = (const float*)d_in[1];
    const float* smu   = (const float*)d_in[2];
    const float* slsig = (const float*)d_in[3];
    const float* niw   = (const float*)d_in[4];
    const float* nib   = (const float*)d_in[5];
    const float* nsw   = (const float*)d_in[6];
    const float* nsb   = (const float*)d_in[7];
    const float* nmw   = (const float*)d_in[8];
    const float* nmb   = (const float*)d_in[9];
    const float* Wq    = (const float*)d_in[10];
    const float* Wk    = (const float*)d_in[11];
    const float* Wv    = (const float*)d_in[12];
    const float* W1    = (const float*)d_in[13];
    const float* b1    = (const float*)d_in[14];
    const float* W2    = (const float*)d_in[15];
    const float* b2    = (const float*)d_in[16];
    const float* pw1   = (const float*)d_in[17];
    const float* pb1   = (const float*)d_in[18];
    const float* pw2   = (const float*)d_in[19];
    const float* pb2   = (const float*)d_in[20];
    const float* tcw   = (const float*)d_in[21];
    const float* tcb   = (const float*)d_in[22];

    char* ws = (char*)d_ws;
    _Float16* packK = (_Float16*)(ws + OFF_PACKWK);
    _Float16* packV = (_Float16*)(ws + OFF_PACKWV);
    _Float16* kbuf  = (_Float16*)(ws + OFF_K);
    _Float16* vbuf  = (_Float16*)(ws + OFF_V);
    float* slots = (float*)(ws + OFF_SLOTS);
    float* qbuf  = (float*)(ws + OFF_Q);
    float* attn  = (float*)(ws + OFF_ATTN);

    repack_weights<<<16, 256, 0, stream>>>(Wk, Wv, packK, packV);
    init_slots<<<(BB * SS * DD + 255) / 256, 256, 0, stream>>>(smu, slsig, noise, slots);
    ln_project<<<1024, 128, 0, stream>>>(obs, niw, nib, packK, packV, kbuf, vbuf);

    for (int it = 0; it < ITERS; ++it) {
        q_proj<<<BB, 128, 0, stream>>>(slots, nsw, nsb, Wq, qbuf);
        attn_softmax<<<dim3(NN / 128, BB), 128, 0, stream>>>(qbuf, kbuf, attn);
        slot_update<<<BB, 128, 0, stream>>>(attn, vbuf, nmw, nmb, W1, b1, W2, b2, slots);
    }
    heads<<<BB, 128, 0, stream>>>(slots, pw1, pb1, pw2, pb2, tcw, tcb, (float*)d_out);
}
